// HierarchicalMemoryCompressor_13932873908444
// MI455X (gfx1250) — matrix operand order repaired and checked by reading, not compiled, not run on hardware
//
#include <hip/hip_runtime.h>

// ---------------------------------------------------------------------------
// HierarchicalMemoryCompressor for MI455X (gfx1250, wave32, WMMA).
// bf16 tensor-core GEMMs (v_wmma_f32_16x16x32_bf16, f32 accumulate).
// v3: global->LDS staging via GLOBAL_LOAD_ASYNC_TO_LDS_B128 (ASYNCcnt,
//     s_wait_asynccnt) instead of load-to-VGPR + ds_store: fewer issue slots,
//     ~40 fewer live VGPRs, DMA overlapped with the 8-WMMA burst.
//     128x64 block tile (32x64 per wave), double-buffered LDS,
//     pre-transposed bf16 weights, global_prefetch of the K+64 slice.
// ---------------------------------------------------------------------------

#define USE_ASYNC_LDS 1   // set 0 to fall back to load+ds_store staging

typedef __bf16 bf16;
typedef __attribute__((ext_vector_type(4)))  __bf16 v4bf;
typedef __attribute__((ext_vector_type(8)))  __bf16 v8bf;
typedef __attribute__((ext_vector_type(16))) __bf16 v16bf;
typedef __attribute__((ext_vector_type(8)))  float  v8f;

#define NTOK  16384   // B*S = 4*4096
#define HDIM  2048
#define LDSS  40      // LDS row stride in halves: 32 data + 8 pad (80B rows)

#if USE_ASYNC_LDS
// 64B (4 x b128) async copy: row contiguous in both global and LDS, so one
// address pair + instruction offsets. Tracked by ASYNCcnt.
__device__ __forceinline__ void async_copy_row64B(unsigned lds_off,
                                                  const void* gptr) {
  asm volatile(
      "global_load_async_to_lds_b128 %0, %1, off\n\t"
      "global_load_async_to_lds_b128 %0, %1, off offset:16\n\t"
      "global_load_async_to_lds_b128 %0, %1, off offset:32\n\t"
      "global_load_async_to_lds_b128 %0, %1, off offset:48"
      :: "v"(lds_off), "v"(gptr) : "memory");
}
__device__ __forceinline__ void async_copy_b128(unsigned lds_off,
                                                const void* gptr) {
  asm volatile("global_load_async_to_lds_b128 %0, %1, off"
               :: "v"(lds_off), "v"(gptr) : "memory");
}
__device__ __forceinline__ void wait_async_zero() {
  asm volatile("s_wait_asynccnt 0x0" ::: "memory");
}
// low 32 bits of a generic LDS pointer == LDS byte offset (ISA aperture rule)
__device__ __forceinline__ unsigned lds_addr32(const void* p) {
  return (unsigned)(size_t)p;
}
#endif

// --------------------------- elementwise kernels ---------------------------

// x -> x_bf16, and combined = [x[:H/2]*freq, x[H/2:]*imp] -> bf16
__global__ __launch_bounds__(256)
void prep_x_kernel(const float* __restrict__ x, const float* __restrict__ freq,
                   const float* __restrict__ imp, bf16* __restrict__ xbf,
                   bf16* __restrict__ comb) {
  size_t idx = ((size_t)blockIdx.x * 256 + threadIdx.x) * 4;
  float4 v = *(const float4*)(x + idx);
  int tok = (int)(idx >> 11);       // / 2048
  int h   = (int)(idx & 2047);
  float m = (h < 1024) ? freq[tok] : imp[tok];   // 1024 % 4 == 0: uniform in vec
  v4bf xb, cb;
  xb[0] = (bf16)v.x;       xb[1] = (bf16)v.y;
  xb[2] = (bf16)v.z;       xb[3] = (bf16)v.w;
  cb[0] = (bf16)(v.x * m); cb[1] = (bf16)(v.y * m);
  cb[2] = (bf16)(v.z * m); cb[3] = (bf16)(v.w * m);
  *(v4bf*)(xbf  + idx) = xb;
  *(v4bf*)(comb + idx) = cb;
}

// out[tok,:] = x[tok,:] where sel[tok]==0
__global__ __launch_bounds__(256)
void copy_sel0_kernel(const float* __restrict__ x, const int* __restrict__ sel,
                      float* __restrict__ out) {
  size_t idx = ((size_t)blockIdx.x * 256 + threadIdx.x) * 4;
  int tok = (int)(idx >> 11);
  if (sel[tok] == 0) {
    *(float4*)(out + idx) = *(const float4*)(x + idx);
  }
}

// wT[n][k] = (bf16) w[k][n]  -- 32x32 LDS tile transpose, f32 -> bf16
__global__ __launch_bounds__(256)
void transpose_f32_to_bf16(const float* __restrict__ w, bf16* __restrict__ wT,
                           int K, int N) {
  __shared__ float tile[32][33];
  const int n0 = blockIdx.x * 32;
  const int k0 = blockIdx.y * 32;
  const int tx = threadIdx.x & 31;
  const int ty = threadIdx.x >> 5;   // 0..7
#pragma unroll
  for (int j = 0; j < 32; j += 8)
    tile[ty + j][tx] = w[(size_t)(k0 + ty + j) * N + n0 + tx];
  __syncthreads();
#pragma unroll
  for (int j = 0; j < 32; j += 8)
    wT[(size_t)(n0 + ty + j) * K + k0 + tx] = (bf16)tile[tx][ty + j];
}

// ------------------------------ WMMA GEMM ----------------------------------
// C[M x N] = act(A[M x K] @ W[K x N] + bias); A row-major bf16, W supplied
// pre-transposed as Wt[N][K] bf16 so both tiles stage as contiguous rows.
// Block: 128 threads (4 waves), 128x64 output tile; wave w: rows w*32..+31,
// all 64 cols = 2 A-frags x 4 B-frags = 8 WMMAs per K-step. K stepped by 32
// through double-buffered LDS fed by async global->LDS DMA: batch for the
// next buffer is issued right after the barrier and completes behind the
// WMMA burst; s_wait_asynccnt 0 at the top of the next step.
// STORE_MODE 0: bf16 dense store. STORE_MODE 1: f32 store where sel==SEL_TGT.
template<bool RELU, int STORE_MODE, int SEL_TGT>
__global__ __launch_bounds__(128)
void gemm_wmma_bf16(const bf16* __restrict__ A, const bf16* __restrict__ Wt,
                    const float* __restrict__ bias, void* __restrict__ outp,
                    const int* __restrict__ sel, int K, int Ncols) {
  __shared__ __align__(16) bf16 ldsA[2][128 * LDSS];
  __shared__ __align__(16) bf16 ldsB[2][64 * LDSS];

  const int tid  = threadIdx.x;
  const int wave = tid >> 5;
  const int lane = tid & 31;
  const int l16  = lane & 15;
  const int lhi  = lane >> 4;
  const int row0 = blockIdx.x * 128;
  const int col0 = blockIdx.y * 64;

  // staging slices: A = 1 row (32 halves) per thread; B = half a row per thread
  const bf16* gA = A  + (size_t)(row0 + tid) * K;
  const bf16* gB = Wt + (size_t)(col0 + (tid >> 1)) * K + (tid & 1) * 16;

  v8f acc[2][4] = {};

#if USE_ASYNC_LDS
  const unsigned ldsA_off[2] = {
      lds_addr32(&ldsA[0][tid * LDSS]), lds_addr32(&ldsA[1][tid * LDSS]) };
  const unsigned ldsB_off[2] = {
      lds_addr32(&ldsB[0][(tid >> 1) * LDSS + (tid & 1) * 16]),
      lds_addr32(&ldsB[1][(tid >> 1) * LDSS + (tid & 1) * 16]) };
  // prologue: DMA k=0 into buffer 0
  async_copy_row64B(ldsA_off[0], gA);
  async_copy_b128(ldsB_off[0], gB);
  async_copy_b128(ldsB_off[0] + 16, gB + 8);
#else
  {
    v8bf a0 = *(const v8bf*)(gA + 0);
    v8bf a1 = *(const v8bf*)(gA + 8);
    v8bf a2 = *(const v8bf*)(gA + 16);
    v8bf a3 = *(const v8bf*)(gA + 24);
    v8bf b0 = *(const v8bf*)(gB);
    *(v8bf*)(&ldsA[0][tid * LDSS + 0])  = a0;
    *(v8bf*)(&ldsA[0][tid * LDSS + 8])  = a1;
    *(v8bf*)(&ldsA[0][tid * LDSS + 16]) = a2;
    *(v8bf*)(&ldsA[0][tid * LDSS + 24]) = a3;
    *(v8bf*)(&ldsB[0][(tid >> 1) * LDSS + (tid & 1) * 16]) = b0;
  }
#endif

  int cur = 0;
  for (int k0 = 0; k0 < K; k0 += 32) {
    const int knxt = k0 + 32;

#if USE_ASYNC_LDS
    wait_async_zero();          // buffer 'cur' DMA complete (this wave)
    __syncthreads();            // all waves done waiting + done reading 'nxt'
    if (knxt < K) {             // kick DMA for 'nxt'; completes behind WMMAs
      const int nxt = cur ^ 1;
      async_copy_row64B(ldsA_off[nxt], gA + knxt);
      async_copy_b128(ldsB_off[nxt], gB + knxt);
      async_copy_b128(ldsB_off[nxt] + 16, gB + knxt + 8);
      if (knxt + 32 < K)
        __builtin_prefetch(gA + knxt + 32, 0, 3);   // global_prefetch_b8
    }
#else
    __syncthreads();
    v8bf na0, na1, na2, na3, nb0;
    if (knxt < K) {
      na0 = *(const v8bf*)(gA + knxt + 0);
      na1 = *(const v8bf*)(gA + knxt + 8);
      na2 = *(const v8bf*)(gA + knxt + 16);
      na3 = *(const v8bf*)(gA + knxt + 24);
      nb0 = *(const v8bf*)(gB + knxt);
      if (knxt + 32 < K)
        __builtin_prefetch(gA + knxt + 32, 0, 3);
    }
#endif

    // ---- A fragments: rows wave*32 + {l16, 16+l16};
    //      lanes 0-15 hold K {0..7,16..23}, lanes 16-31 K {8..15,24..31} ----
    v16bf af[2];
    {
      const bf16* pa0 = &ldsA[cur][(wave * 32 + l16) * LDSS + lhi * 8];
      const bf16* pa1 = pa0 + 16 * LDSS;
      v8bf lo0 = *(const v8bf*)(pa0);  v8bf hi0 = *(const v8bf*)(pa0 + 16);
      v8bf lo1 = *(const v8bf*)(pa1);  v8bf hi1 = *(const v8bf*)(pa1 + 16);
#pragma unroll
      for (int j = 0; j < 8; ++j) {
        af[0][j] = lo0[j]; af[0][8 + j] = hi0[j];
        af[1][j] = lo1[j]; af[1][8 + j] = hi1[j];
      }
    }
    // ---- B fragments: lane = col, lanes 0-15 K {0..7,16..23}, lanes 16-31 K {8..15,24..31} ----
    v16bf bfr[4];
#pragma unroll
    for (int s = 0; s < 4; ++s) {
      const bf16* pb = &ldsB[cur][(s * 16 + l16) * LDSS + lhi * 8];
      v8bf lo = *(const v8bf*)(pb);  v8bf hi = *(const v8bf*)(pb + 16);
#pragma unroll
      for (int j = 0; j < 8; ++j) { bfr[s][j] = lo[j]; bfr[s][8 + j] = hi[j]; }
    }

    // ---- 8 back-to-back WMMAs ----
#pragma unroll
    for (int s = 0; s < 4; ++s) {
      acc[0][s] = __builtin_amdgcn_wmma_f32_16x16x32_bf16(
          false, af[0], false, bfr[s], (short)0, acc[0][s], false, false);
      acc[1][s] = __builtin_amdgcn_wmma_f32_16x16x32_bf16(
          false, af[1], false, bfr[s], (short)0, acc[1][s], false, false);
    }

#if !USE_ASYNC_LDS
    if (knxt < K) {
      const int nxt = cur ^ 1;
      *(v8bf*)(&ldsA[nxt][tid * LDSS + 0])  = na0;
      *(v8bf*)(&ldsA[nxt][tid * LDSS + 8])  = na1;
      *(v8bf*)(&ldsA[nxt][tid * LDSS + 16]) = na2;
      *(v8bf*)(&ldsA[nxt][tid * LDSS + 24]) = na3;
      *(v8bf*)(&ldsB[nxt][(tid >> 1) * LDSS + (tid & 1) * 16]) = nb0;
    }
#endif
    cur ^= 1;
  }

  // ---- epilogue: C/D layout lanes 0-15: M=v, N=lane; lanes 16-31: M=v+8 ----
#pragma unroll
  for (int i = 0; i < 2; ++i) {
    const int rbase = row0 + wave * 32 + i * 16 + lhi * 8;
#pragma unroll
    for (int s = 0; s < 4; ++s) {
      const int col = col0 + s * 16 + l16;
      const float bv = bias[col];
#pragma unroll
      for (int v = 0; v < 8; ++v) {
        float val = acc[i][s][v] + bv;
        if (RELU) val = val > 0.f ? val : 0.f;
        const int row = rbase + v;
        if (STORE_MODE == 0) {
          ((bf16*)outp)[(size_t)row * Ncols + col] = (bf16)val;
        } else {
          if (sel[row] == SEL_TGT)
            ((float*)outp)[(size_t)row * Ncols + col] = val;
        }
      }
    }
  }
}

// --------------------- selector logits (K=3) + argmax ----------------------
// One wave per token; softmax is monotone so argmax(logits) == argmax(softmax).
__global__ __launch_bounds__(256)
void selector_argmax_kernel(const bf16* __restrict__ hmid,
                            const float* __restrict__ W2,
                            const float* __restrict__ b2,
                            int* __restrict__ sel) {
  const int wave = threadIdx.x >> 5;
  const int lane = threadIdx.x & 31;
  const int tok  = blockIdx.x * 8 + wave;
  const bf16* h = hmid + (size_t)tok * 512;
  float a0 = 0.f, a1 = 0.f, a2 = 0.f;
#pragma unroll
  for (int j = lane; j < 512; j += 32) {
    float hv = (float)h[j];
    a0 += hv * W2[j * 3 + 0];
    a1 += hv * W2[j * 3 + 1];
    a2 += hv * W2[j * 3 + 2];
  }
#pragma unroll
  for (int off = 16; off > 0; off >>= 1) {
    a0 += __shfl_down(a0, off, 32);
    a1 += __shfl_down(a1, off, 32);
    a2 += __shfl_down(a2, off, 32);
  }
  if (lane == 0) {
    a0 += b2[0]; a1 += b2[1]; a2 += b2[2];
    int best = 0; float bv = a0;
    if (a1 > bv) { bv = a1; best = 1; }   // strict > matches argmax tie-break
    if (a2 > bv) { best = 2; }
    sel[tok] = best;
  }
}

// ------------------------------- launcher ----------------------------------

extern "C" void kernel_launch(void* const* d_in, const int* in_sizes, int n_in,
                              void* d_out, int out_size, void* d_ws, size_t ws_size,
                              hipStream_t stream) {
  const float* x      = (const float*)d_in[0];
  const float* freq   = (const float*)d_in[1];
  const float* imp    = (const float*)d_in[2];
  const float* sel_W1 = (const float*)d_in[3];
  const float* sel_b1 = (const float*)d_in[4];
  const float* sel_W2 = (const float*)d_in[5];
  const float* sel_b2 = (const float*)d_in[6];
  const float* c1_W   = (const float*)d_in[7];
  const float* c1_b   = (const float*)d_in[8];
  const float* f1_W   = (const float*)d_in[9];
  const float* f1_b   = (const float*)d_in[10];
  const float* d1_W   = (const float*)d_in[11];
  const float* d1_b   = (const float*)d_in[12];
  const float* c2_W   = (const float*)d_in[13];
  const float* c2_b   = (const float*)d_in[14];
  const float* f2_W   = (const float*)d_in[15];
  const float* f2_b   = (const float*)d_in[16];
  const float* d2_W   = (const float*)d_in[17];
  const float* d2_b   = (const float*)d_in[18];

  // -------- workspace layout (~177 MB, sequential buffer reuse) ------------
  // Weight buffers hold bf16 *transposed* weights: wXT[N][K].
  char* p = (char*)d_ws;
  bf16* wSel1T = (bf16*)p; p += (size_t)2048 * 512  * 2;
  bf16* wC1T   = (bf16*)p; p += (size_t)2048 * 1024 * 2;
  bf16* wF1T   = (bf16*)p; p += (size_t)1024 * 1024 * 2;
  bf16* wD1T   = (bf16*)p; p += (size_t)1024 * 2048 * 2;
  bf16* wC2T   = (bf16*)p; p += (size_t)2048 * 512  * 2;
  bf16* wF2T   = (bf16*)p; p += (size_t)512  * 512  * 2;
  bf16* wD2T   = (bf16*)p; p += (size_t)512  * 2048 * 2;
  bf16* xbf    = (bf16*)p; p += (size_t)NTOK * 2048 * 2;
  bf16* comb   = (bf16*)p;                      // reused as t1/t1b after GEMM1
  bf16* t1     = comb;
  bf16* t1b    = comb + (size_t)NTOK * 1024;
  p += (size_t)NTOK * 2048 * 2;
  bf16* hmid   = (bf16*)p;                      // reused as t2/t2b after argmax
  bf16* t2     = hmid;
  bf16* t2b    = hmid + (size_t)NTOK * 512;
  p += (size_t)NTOK * 1024 * 2;
  int*  sel    = (int*)p;

  // -------- 1) weight f32 -> bf16 with transpose ---------------------------
  auto cvtT = [&](const float* s, bf16* d, int K, int N) {
    transpose_f32_to_bf16<<<dim3(N / 32, K / 32), 256, 0, stream>>>(s, d, K, N);
  };
  cvtT(sel_W1, wSel1T, 2048, 512);
  cvtT(c1_W,   wC1T,   2048, 1024);
  cvtT(f1_W,   wF1T,   1024, 1024);
  cvtT(d1_W,   wD1T,   1024, 2048);
  cvtT(c2_W,   wC2T,   2048, 512);
  cvtT(f2_W,   wF2T,   512,  512);
  cvtT(d2_W,   wD2T,   512,  2048);

  // -------- 2) activations: x_bf16 and modulated 'combined' ----------------
  prep_x_kernel<<<(NTOK * HDIM / 4) / 256, 256, 0, stream>>>(x, freq, imp, xbf, comb);

  const dim3 blk(128);
  const unsigned MB = NTOK / 128;   // 128 row-blocks

  // -------- 3) selector: hmid = relu(combined @ sel_W1 + b1) ---------------
  gemm_wmma_bf16<true, 0, 0><<<dim3(MB, 512 / 64), blk, 0, stream>>>(
      comb, wSel1T, sel_b1, hmid, nullptr, 2048, 512);
  // -------- 4) logits + argmax -> sel --------------------------------------
  selector_argmax_kernel<<<NTOK / 8, 256, 0, stream>>>(hmid, sel_W2, sel_b2, sel);

  // -------- 5) expert 1 chain ----------------------------------------------
  gemm_wmma_bf16<false, 0, 0><<<dim3(MB, 1024 / 64), blk, 0, stream>>>(
      xbf, wC1T, c1_b, t1, nullptr, 2048, 1024);
  gemm_wmma_bf16<false, 0, 0><<<dim3(MB, 1024 / 64), blk, 0, stream>>>(
      t1, wF1T, f1_b, t1b, nullptr, 1024, 1024);
  gemm_wmma_bf16<false, 1, 1><<<dim3(MB, 2048 / 64), blk, 0, stream>>>(
      t1b, wD1T, d1_b, d_out, sel, 1024, 2048);

  // -------- 6) expert 2 chain ----------------------------------------------
  gemm_wmma_bf16<false, 0, 0><<<dim3(MB, 512 / 64), blk, 0, stream>>>(
      xbf, wC2T, c2_b, t2, nullptr, 2048, 512);
  gemm_wmma_bf16<false, 0, 0><<<dim3(MB, 512 / 64), blk, 0, stream>>>(
      t2, wF2T, f2_b, t2b, nullptr, 512, 512);
  gemm_wmma_bf16<false, 1, 2><<<dim3(MB, 2048 / 64), blk, 0, stream>>>(
      t2b, wD2T, d2_b, d_out, sel, 512, 2048);

  // -------- 7) identity path (sel==0): out = x -----------------------------
  copy_sel0_kernel<<<(NTOK * HDIM / 4) / 256, 256, 0, stream>>>(x, sel, (float*)d_out);
}
